// MultiHeadAttention_12249246728908
// MI455X (gfx1250) — matrix operand order repaired and checked by reading, not compiled, not run on hardware
//
#include <hip/hip_runtime.h>
#include <hip/hip_bf16.h>

// MHA for B=2, S=2048, D=1024, H=16, DH=64 on gfx1250 (wave32, WMMA).
// f16 WMMA (v_wmma_f32_16x16x32_f16) with f32 accumulate everywhere.

typedef __attribute__((ext_vector_type(16))) _Float16 v16h;
typedef __attribute__((ext_vector_type(8)))  _Float16 v8h;
typedef __attribute__((ext_vector_type(8)))  float    v8f;

#define DEV __device__ __forceinline__

static constexpr int BB = 2, SS = 2048, DD = 1024, HH = 16, DH = 64;

DEV v16h cmb(v8h lo, v8h hi) {
  return __builtin_shufflevector(lo, hi, 0,1,2,3,4,5,6,7,8,9,10,11,12,13,14,15);
}
// 16 contiguous halves (two 16B chunks) -> B-fragment / generic frag
DEV v16h load_frag16(const _Float16* p) {
  v8h lo = *(const v8h*)(p);
  v8h hi = *(const v8h*)(p + 8);
  return cmb(lo, hi);
}
// A-fragment (16x32 f16): lane row-major; a[0..7]=row[kb+8h .. +7], a[8..15]=row[kb+16+8h ..]
DEV v16h load_afrag(const _Float16* rowp, int kbase, int half) {
  v8h lo = *(const v8h*)(rowp + kbase + 8 * half);
  v8h hi = *(const v8h*)(rowp + kbase + 16 + 8 * half);
  return cmb(lo, hi);
}
DEV v8f wmma16(v16h a, v16h b, v8f c) {
  return __builtin_amdgcn_wmma_f32_16x16x32_f16(false, a, false, b, (short)0, c, false, false);
}

// Single-instruction butterfly stages: DPP ROW_XMASK fused onto the ALU op
// (d = op(dpp_xor(v), v)); avoids the mov + canonicalize pair.
#define DPP_RED_OP(NAME, OP, MASK)                                           \
  DEV float NAME(float v) {                                                  \
    float d;                                                                 \
    asm(OP " %0, %1, %1 row_xmask:" #MASK                                    \
        " row_mask:0xf bank_mask:0xf bound_ctrl:1"                           \
        : "=v"(d) : "v"(v));                                                 \
    return d;                                                                \
  }
DPP_RED_OP(fmax_x1, "v_max_num_f32_dpp", 1)
DPP_RED_OP(fmax_x2, "v_max_num_f32_dpp", 2)
DPP_RED_OP(fmax_x4, "v_max_num_f32_dpp", 4)
DPP_RED_OP(fmax_x8, "v_max_num_f32_dpp", 8)
DPP_RED_OP(fadd_x1, "v_add_f32_dpp", 1)
DPP_RED_OP(fadd_x2, "v_add_f32_dpp", 2)
DPP_RED_OP(fadd_x4, "v_add_f32_dpp", 4)
DPP_RED_OP(fadd_x8, "v_add_f32_dpp", 8)

// reductions across the 16-lane N-group of the C/D layout
DEV float redmax16(float v) {
  v = fmax_x1(v); v = fmax_x2(v); v = fmax_x4(v); v = fmax_x8(v);
  return v;
}
DEV float redsum16(float v) {
  v = fadd_x1(v); v = fadd_x2(v); v = fadd_x4(v); v = fadd_x8(v);
  return v;
}

// ---------------------------------------------------------------------------
// Projection GEMM: C[M=4096, N=1024] = A[M,1024] (f32) * W (+bias), K=1024.
// MODE 0: W=[H,D,DH] per-head, out f16 [b,h,s,64]      (Q, K projections)
// MODE 1: W=[H,D,DH] per-head, out f16 [b,h,64,s]      (V projection, transposed)
// MODE 2: W=[D,D] plain,       out f32 [B,S,D]         (output projection)
// Block: 256 threads (8 waves), tile 128(M) x 64(N), K-step 32.
// ---------------------------------------------------------------------------
template <int MODE>
__global__ void __launch_bounds__(256)
proj_gemm(const float* __restrict__ A, const float* __restrict__ W,
          const float* __restrict__ bias, void* __restrict__ Out) {
  __shared__ alignas(32) _Float16 lds_a[128][40];  // 128 x 32 (+8 pad)
  __shared__ alignas(32) _Float16 lds_bt[64][40];  // W^T tile: [n][k] 64 x 32 (+8 pad)

  const int tid  = threadIdx.x;
  const int wv   = tid >> 5;
  const int lane = tid & 31;
  const int l16  = lane & 15;
  const int half = lane >> 4;
  const int m0 = blockIdx.x * 128;
  const int n0 = blockIdx.y * 64;

  v8f acc[4] = {};

  for (int k0 = 0; k0 < DD; k0 += 32) {
    // stage A tile 128x32 f32 -> f16 (two 16B vector stores per thread)
    {
      const int row = tid >> 1, cb = (tid & 1) * 16;
      const float* src = A + (size_t)(m0 + row) * DD + k0 + cb;
      v8h h0, h1;
      #pragma unroll
      for (int i = 0; i < 8; ++i) { h0[i] = (_Float16)src[i]; h1[i] = (_Float16)src[8 + i]; }
      *(v8h*)&lds_a[row][cb]     = h0;
      *(v8h*)&lds_a[row][cb + 8] = h1;
    }
    // stage W tile 32(k) x 64(n), transposed into lds_bt[n][k]
    {
      const int kd = tid >> 3, nb = (tid & 7) * 8;
      const float* src;
      if (MODE == 2)
        src = W + (size_t)(k0 + kd) * DD + n0 + nb;                 // Wo[d][n]
      else
        src = W + ((size_t)blockIdx.y * DD + (k0 + kd)) * DH + nb;  // W[h][d][dh]
      #pragma unroll
      for (int i = 0; i < 8; ++i) lds_bt[nb + i][kd] = (_Float16)src[i];
    }
    __syncthreads();

    const _Float16* ar = &lds_a[wv * 16 + l16][0];
    const v16h af = cmb(*(const v8h*)(ar + 8 * half),
                        *(const v8h*)(ar + 16 + 8 * half));
    #pragma unroll
    for (int t = 0; t < 4; ++t) {
      const _Float16* br = &lds_bt[t * 16 + l16][0];
      acc[t] = wmma16(af, load_afrag(br, 0, half), acc[t]);
    }
    __syncthreads();
  }

  // epilogue: bias + store in target layout
  #pragma unroll
  for (int t = 0; t < 4; ++t) {
    const int col = t * 16 + l16;
    #pragma unroll
    for (int r = 0; r < 8; ++r) {
      const int mrow = m0 + wv * 16 + r + 8 * half;
      const float val = acc[t][r] + bias[n0 + col];
      if (MODE == 2) {
        ((float*)Out)[(size_t)mrow * DD + n0 + col] = val;
      } else {
        const int bb = mrow >> 11;         // mrow / S
        const int s  = mrow & (SS - 1);
        _Float16* o = (_Float16*)Out;
        const int hh = blockIdx.y;         // N-tile == head (DH == 64)
        if (MODE == 0)
          o[(((size_t)(bb * HH + hh)) * SS + s) * DH + col] = (_Float16)val;
        else
          o[(((size_t)(bb * HH + hh)) * DH + col) * SS + s] = (_Float16)val;
      }
    }
  }
}

// ---------------------------------------------------------------------------
// Attention: one wave handles 16 query rows for a fixed (b,h).
// Pass 1: flash online softmax (exp2 domain) + ctx accumulation (aw@V WMMA).
// Pass 2: recompute scores, write normalized aws once.
// ---------------------------------------------------------------------------
__global__ void __launch_bounds__(256)
attn_kernel(const _Float16* __restrict__ Qh, const _Float16* __restrict__ Kh,
            const _Float16* __restrict__ Vt, float* __restrict__ ctx,
            float* __restrict__ aw) {
  __shared__ alignas(32) _Float16 pb[8][16][40];  // per-wave p tile (16 q x 32 keys)

  const int tid  = threadIdx.x;
  const int wv   = tid >> 5;
  const int lane = tid & 31;
  const int l16  = lane & 15;
  const int half = lane >> 4;

  const int wg = blockIdx.x * 8 + wv;        // 0..4095
  const int qt = wg & (SS / 16 - 1);         // query tile (128)
  const int h  = (wg >> 7) & (HH - 1);
  const int b  = wg >> 11;
  const int q0 = qt * 16;

  const _Float16* qrow = Qh + (((size_t)(b * HH + h)) * SS + q0 + l16) * DH;
  const v16h aq0 = load_afrag(qrow, 0, half);    // Q, d = 0..31
  const v16h aq1 = load_afrag(qrow, 32, half);   // Q, d = 32..63
  const _Float16* kb = Kh + ((size_t)(b * HH + h)) * SS * DH;   // [key][d]
  const _Float16* vb = Vt + ((size_t)(b * HH + h)) * DH * SS;   // [dim][key]

  float m[8], l[8];
  v8f acc[4] = {};
  #pragma unroll
  for (int r = 0; r < 8; ++r) { m[r] = -3.0e38f; l[r] = 0.0f; }
  // exp2-domain scale: (1/sqrt(DH)) * log2(e); softmax identical under base change
  const float csc = 0.18033688011112042f;

  for (int k0 = 0; k0 < SS; k0 += 32) {
    const _Float16* kr0 = kb + (size_t)(k0 + l16) * DH;  // my key column (keys k0..k0+15)
    const _Float16* kr1 = kr0 + 16 * DH;                 // keys k0+16..k0+31
    v8f s0 = {}, s1 = {};
    s0 = wmma16(aq0, load_afrag(kr0, 0, half), s0);
    s0 = wmma16(aq1, load_afrag(kr0, 32, half), s0);
    s1 = wmma16(aq0, load_afrag(kr1, 0, half), s1);
    s1 = wmma16(aq1, load_afrag(kr1, 32, half), s1);

    #pragma unroll
    for (int r = 0; r < 8; ++r) {
      const float a0 = s0[r] * csc, a1 = s1[r] * csc;
      const float mx = redmax16(fmaxf(a0, a1));
      const float mn = fmaxf(m[r], mx);
      const float alpha = __builtin_amdgcn_exp2f(m[r] - mn);
      const float p0 = __builtin_amdgcn_exp2f(a0 - mn);
      const float p1 = __builtin_amdgcn_exp2f(a1 - mn);
      l[r] = l[r] * alpha + redsum16(p0 + p1);
      m[r] = mn;
      acc[0][r] *= alpha; acc[1][r] *= alpha;
      acc[2][r] *= alpha; acc[3][r] *= alpha;
      // C-layout -> LDS (row-major p tile)
      pb[wv][r + 8 * half][l16]      = (_Float16)p0;
      pb[wv][r + 8 * half][16 + l16] = (_Float16)p1;
    }
    asm volatile("s_wait_dscnt 0" ::: "memory");  // wave-local LDS re-stripe fence
    const v16h pf = load_afrag(&pb[wv][l16][0], 0, half);  // A-layout of p
    #pragma unroll
    for (int t = 0; t < 4; ++t) {
      const _Float16* vr = vb + (size_t)(t * 16 + l16) * SS + k0;
      acc[t] = wmma16(pf, load_afrag(vr, 0, half), acc[t]);
    }
  }

  #pragma unroll
  for (int r = 0; r < 8; ++r) l[r] = 1.0f / l[r];

  // ctx: [B,S,D] f32, column = h*64 + dim
  #pragma unroll
  for (int t = 0; t < 4; ++t) {
    #pragma unroll
    for (int r = 0; r < 8; ++r) {
      ctx[((size_t)b * SS + q0 + r + 8 * half) * DD + h * DH + t * 16 + l16] =
          acc[t][r] * l[r];
    }
  }

  // Pass 2: recompute scores, emit normalized attention weights [H,B,S,S]
  float* awb = aw + (((size_t)(h * BB + b)) * SS + q0) * SS;
  for (int k0 = 0; k0 < SS; k0 += 32) {
    const _Float16* kr0 = kb + (size_t)(k0 + l16) * DH;
    const _Float16* kr1 = kr0 + 16 * DH;
    v8f s0 = {}, s1 = {};
    s0 = wmma16(aq0, load_afrag(kr0, 0, half), s0);
    s0 = wmma16(aq1, load_afrag(kr0, 32, half), s0);
    s1 = wmma16(aq0, load_afrag(kr1, 0, half), s1);
    s1 = wmma16(aq1, load_afrag(kr1, 32, half), s1);
    #pragma unroll
    for (int r = 0; r < 8; ++r) {
      const float p0 = __builtin_amdgcn_exp2f(s0[r] * csc - m[r]) * l[r];
      const float p1 = __builtin_amdgcn_exp2f(s1[r] * csc - m[r]) * l[r];
      float* orow = awb + (size_t)(r + 8 * half) * SS + k0 + l16;
      orow[0]  = p0;
      orow[16] = p1;
    }
  }
}

// ---------------------------------------------------------------------------
extern "C" void kernel_launch(void* const* d_in, const int* in_sizes, int n_in,
                              void* d_out, int out_size, void* d_ws, size_t ws_size,
                              hipStream_t stream) {
  const float* q  = (const float*)d_in[0];
  const float* k  = (const float*)d_in[1];
  const float* v  = (const float*)d_in[2];
  const float* Wq = (const float*)d_in[3];
  const float* Wk = (const float*)d_in[4];
  const float* Wv = (const float*)d_in[5];
  const float* bq = (const float*)d_in[6];
  const float* bk = (const float*)d_in[7];
  const float* bv = (const float*)d_in[8];
  const float* Wo = (const float*)d_in[9];
  const float* bo = (const float*)d_in[10];

  char* ws = (char*)d_ws;
  const size_t QKV_BYTES = (size_t)BB * HH * SS * DH * sizeof(_Float16);  // 8 MB
  _Float16* Qh = (_Float16*)(ws);
  _Float16* Kh = (_Float16*)(ws + QKV_BYTES);
  _Float16* Vt = (_Float16*)(ws + 2 * QKV_BYTES);
  float*    cx = (float*)(ws + 3 * QKV_BYTES);   // ctx f32, 16 MB (total 40 MB)

  float* out = (float*)d_out;                                // [B,S,D]
  float* aws = out + (size_t)BB * SS * DD;                   // [H,B,S,S]

  dim3 grid(32, 16), blk(256);
  proj_gemm<0><<<grid, blk, 0, stream>>>(q, Wq, bq, Qh);
  proj_gemm<0><<<grid, blk, 0, stream>>>(k, Wk, bk, Kh);
  proj_gemm<1><<<grid, blk, 0, stream>>>(v, Wv, bv, Vt);
  attn_kernel<<<dim3(512), blk, 0, stream>>>(Qh, Kh, Vt, cx, aws);
  proj_gemm<2><<<grid, blk, 0, stream>>>(cx, Wo, bo, out);
}
